// TinySeq2Seq_88278757802440
// MI455X (gfx1250) — matrix operand order repaired and checked by reading, not compiled, not run on hardware
//
#include <hip/hip_runtime.h>
#include <hip/hip_bf16.h>

#define HSZ   512
#define VTOT  32001
#define EOSID 32000
#define GSZ   2048          // 4*H
#define VPAD  32016         // 2001 tiles of 16 rows
#define NTILE 2001

typedef __attribute__((ext_vector_type(16))) __bf16 v16bf;
typedef __attribute__((ext_vector_type(8)))  float  v8f;

union BfVec { uint4 q[2]; v16bf v; unsigned short s[16]; };

__device__ __forceinline__ unsigned short f2bf(float f) {
    unsigned int u = __builtin_bit_cast(unsigned int, f);
    unsigned int r = u + 0x7FFFu + ((u >> 16) & 1u);
    return (unsigned short)(r >> 16);
}

__device__ __forceinline__ float sigm(float x) { return 1.0f / (1.0f + __expf(-x)); }

// A-matrix fragment: 16-bit A 16x32 layout. Lane m(0..15) holds row m:
// K = kc*32 + {0..7, 16..23}; lanes 16..31 hold K = {8..15, 24..31}.
// p points at element (row*512 + kc*32 + khalf*8): chunks at +0 and +16 elems.
__device__ __forceinline__ v16bf load_a16(const unsigned short* p) {
    BfVec u;
    u.q[0] = ((const uint4*)p)[0];   // +0  .. +7
    u.q[1] = ((const uint4*)p)[2];   // +16 .. +23
    return u.v;
}
// B-matrix fragment (broadcast h chunk into all 16 columns):
// lanes 0..15 carry K = kc*32 + {0..7, 16..23}, lanes 16..31 carry K = kc*32 + {8..15, 24..31}.
__device__ __forceinline__ v16bf load_b16(const unsigned short* p) {
    BfVec u;
    u.q[0] = ((const uint4*)p)[0];
    u.q[1] = ((const uint4*)p)[2];
    return u.v;
}

// ---- one-time bf16 conversion of enc_Whh, dec_Whh, proj_W (padded to 32016 rows) ----
__global__ void convert_kernel(const float* __restrict__ encWhh,
                               const float* __restrict__ decWhh,
                               const float* __restrict__ projW,
                               unsigned short* __restrict__ encBf,
                               unsigned short* __restrict__ decBf,
                               unsigned short* __restrict__ projBf) {
    const long NW = (long)GSZ * HSZ;          // 1,048,576
    const long NP = (long)VPAD * HSZ;         // 16,392,192
    const long total = NP + 2 * NW;
    for (long i = (long)blockIdx.x * blockDim.x + threadIdx.x; i < total;
         i += (long)gridDim.x * blockDim.x) {
        if (i < NP) {
            long row = i >> 9;
            long col = i & 511;
            float v = (row < VTOT) ? projW[row * HSZ + col] : 0.0f;
            projBf[i] = f2bf(v);
        } else if (i < NP + NW) {
            long j = i - NP;       encBf[j] = f2bf(encWhh[j]);
        } else {
            long j = i - NP - NW;  decBf[j] = f2bf(decWhh[j]);
        }
    }
}

// ---- xg[s][g] = enc_Wih[g, src[s]] + enc_bih[g]  (column gather, all 128 steps) ----
__global__ void xg_kernel(const float* __restrict__ encWih,
                          const float* __restrict__ bih,
                          const int* __restrict__ src,
                          float* __restrict__ xg) {
    int i = blockIdx.x * blockDim.x + threadIdx.x;
    if (i >= 128 * GSZ) return;
    int s = i >> 11;
    int g = i & (GSZ - 1);
    xg[i] = encWih[(size_t)g * VTOT + src[s]] + bih[g];
}

// ---- persistent encoder: 128 LSTM steps in one WGP, Whh@h via WMMA bf16 ----
__global__ void __launch_bounds__(256)
enc_kernel(const float* __restrict__ xg,
           const unsigned short* __restrict__ WhhBf,
           const float* __restrict__ bhh,
           float* __restrict__ cWs, unsigned short* __restrict__ hbfWs,
           int* __restrict__ state) {
    __shared__ float c_s[HSZ];
    __shared__ float g_s[GSZ];
    __shared__ __align__(32) unsigned short hbf_s[HSZ];
    const int tid = threadIdx.x;
    for (int e = tid; e < HSZ; e += 256) { c_s[e] = 0.0f; hbf_s[e] = 0; }
    __syncthreads();
    const int wave = tid >> 5, lane = tid & 31;
    const int mrow = lane & 15, khalf = lane >> 4;
    for (int t = 0; t < 128; ++t) {
        // gates = Whh @ h : 128 row-tiles of 16, 8 waves, 16 tiles each
        for (int tile = wave; tile < 128; tile += 8) {
            const int rowBase = tile << 4;
            v8f acc = {0.f, 0.f, 0.f, 0.f, 0.f, 0.f, 0.f, 0.f};
            for (int kc = 0; kc < 16; ++kc) {
                v16bf a = load_a16(WhhBf + (size_t)(rowBase + mrow) * HSZ + kc * 32 + khalf * 8);
                v16bf b = load_b16(hbf_s + kc * 32 + (khalf << 3));
                acc = __builtin_amdgcn_wmma_f32_16x16x32_bf16(
                          false, a, false, b, (short)0, acc, false, false);
            }
            if (lane == 0)       for (int j = 0; j < 8; ++j) g_s[rowBase + j] = acc[j];
            else if (lane == 16) for (int j = 0; j < 8; ++j) g_s[rowBase + 8 + j] = acc[j];
        }
        __syncthreads();
        const float* xgt = xg + (size_t)t * GSZ;
        for (int e = tid; e < HSZ; e += 256) {
            float Gi = g_s[e]            + xgt[e]            + bhh[e];
            float Gf = g_s[e + 512]      + xgt[e + 512]      + bhh[e + 512];
            float Gg = g_s[e + 1024]     + xgt[e + 1024]     + bhh[e + 1024];
            float Go = g_s[e + 1536]     + xgt[e + 1536]     + bhh[e + 1536];
            float i_ = sigm(Gi), f_ = sigm(Gf), o_ = sigm(Go), gg = tanhf(Gg);
            float cn = f_ * c_s[e] + i_ * gg;
            c_s[e] = cn;
            hbf_s[e] = f2bf(o_ * tanhf(cn));
        }
        __syncthreads();
    }
    for (int e = tid; e < HSZ; e += 256) { cWs[e] = c_s[e]; hbfWs[e] = hbf_s[e]; }
    if (tid == 0) { state[0] = -1; state[1] = 0; }   // idx = -1, done = 0
}

// ---- decoder LSTM cell for one step (freezes state when done) ----
__global__ void __launch_bounds__(256)
dec_cell_kernel(const unsigned short* __restrict__ WhhBf,
                const float* __restrict__ decWih,
                const float* __restrict__ bih, const float* __restrict__ bhh,
                int* __restrict__ state,
                float* __restrict__ cWs, unsigned short* __restrict__ hbfWs) {
    __shared__ float g_s[GSZ];
    const int tid = threadIdx.x;
    const int idx = state[0], done = state[1];
    const int wave = tid >> 5, lane = tid & 31;
    const int mrow = lane & 15, khalf = lane >> 4;
    for (int tile = wave; tile < 128; tile += 8) {
        const int rowBase = tile << 4;
        v8f acc = {0.f, 0.f, 0.f, 0.f, 0.f, 0.f, 0.f, 0.f};
        for (int kc = 0; kc < 16; ++kc) {
            v16bf a = load_a16(WhhBf + (size_t)(rowBase + mrow) * HSZ + kc * 32 + khalf * 8);
            v16bf b = load_b16(hbfWs + kc * 32 + (khalf << 3));
            acc = __builtin_amdgcn_wmma_f32_16x16x32_bf16(
                      false, a, false, b, (short)0, acc, false, false);
        }
        if (lane == 0)       for (int j = 0; j < 8; ++j) g_s[rowBase + j] = acc[j];
        else if (lane == 16) for (int j = 0; j < 8; ++j) g_s[rowBase + 8 + j] = acc[j];
    }
    __syncthreads();
    for (int e = tid; e < HSZ; e += 256) {
        float xi = 0.f, xf = 0.f, xc = 0.f, xo = 0.f;
        if (idx >= 0) {
            xi = decWih[(size_t)(e)        * VTOT + idx];
            xf = decWih[(size_t)(e + 512)  * VTOT + idx];
            xc = decWih[(size_t)(e + 1024) * VTOT + idx];
            xo = decWih[(size_t)(e + 1536) * VTOT + idx];
        }
        float Gi = g_s[e]        + xi + bih[e]        + bhh[e];
        float Gf = g_s[e + 512]  + xf + bih[e + 512]  + bhh[e + 512];
        float Gg = g_s[e + 1024] + xc + bih[e + 1024] + bhh[e + 1024];
        float Go = g_s[e + 1536] + xo + bih[e + 1536] + bhh[e + 1536];
        float i_ = sigm(Gi), f_ = sigm(Gf), o_ = sigm(Go), gg = tanhf(Gg);
        float cn = f_ * cWs[e] + i_ * gg;
        float hn = o_ * tanhf(cn);
        if (!done) { cWs[e] = cn; hbfWs[e] = f2bf(hn); }
    }
}

// ---- proj_W @ h + b: WMMA matvec across 64 WGPs, logits out + per-block argmax ----
__global__ void __launch_bounds__(256)
proj_kernel(const unsigned short* __restrict__ projBf,
            const unsigned short* __restrict__ hbfWs,
            const float* __restrict__ projB,
            const int* __restrict__ state,
            float* __restrict__ outLogits,
            float* __restrict__ partVal, int* __restrict__ partIdx) {
    __shared__ float bv[256];
    __shared__ int   bi[256];
    const int tid = threadIdx.x;
    const int done = state[1];
    const int wave = tid >> 5, lane = tid & 31;
    const int gw = blockIdx.x * 8 + wave;              // 0..511
    const int mrow = lane & 15, khalf = lane >> 4;
    float bestv = -3.4e38f; int besti = 0x7fffffff;
    for (int tile = gw; tile < NTILE; tile += 512) {
        const int rowBase = tile << 4;
        v8f acc = {0.f, 0.f, 0.f, 0.f, 0.f, 0.f, 0.f, 0.f};
        for (int kc = 0; kc < 16; ++kc) {
            v16bf a = load_a16(projBf + (size_t)(rowBase + mrow) * HSZ + kc * 32 + khalf * 8);
            v16bf b = load_b16(hbfWs + kc * 32 + (khalf << 3));
            acc = __builtin_amdgcn_wmma_f32_16x16x32_bf16(
                      false, a, false, b, (short)0, acc, false, false);
        }
        const int half = (lane == 0) ? 0 : ((lane == 16) ? 1 : -1);
        if (half >= 0) {
            for (int j = 0; j < 8; ++j) {
                int r = rowBase + half * 8 + j;
                if (r < VTOT) {
                    float lg = acc[j] + projB[r];
                    outLogits[r] = done ? 0.0f : lg;
                    if (lg > bestv || (lg == bestv && r < besti)) { bestv = lg; besti = r; }
                }
            }
        }
    }
    bv[tid] = bestv; bi[tid] = besti;
    __syncthreads();
    for (int s = 128; s > 0; s >>= 1) {
        if (tid < s) {
            float v2 = bv[tid + s]; int i2 = bi[tid + s];
            if (v2 > bv[tid] || (v2 == bv[tid] && i2 < bi[tid])) { bv[tid] = v2; bi[tid] = i2; }
        }
        __syncthreads();
    }
    if (tid == 0) { partVal[blockIdx.x] = bv[0]; partIdx[blockIdx.x] = bi[0]; }
}

// ---- final argmax over 64 partials + greedy-decode state update ----
__global__ void argmax_kernel(const float* __restrict__ partVal,
                              const int* __restrict__ partIdx,
                              int* __restrict__ state,
                              float* __restrict__ outIdx) {
    __shared__ float bv[64];
    __shared__ int   bi[64];
    const int tid = threadIdx.x;
    bv[tid] = partVal[tid]; bi[tid] = partIdx[tid];
    __syncthreads();
    for (int s = 32; s > 0; s >>= 1) {
        if (tid < s) {
            float v2 = bv[tid + s]; int i2 = bi[tid + s];
            if (v2 > bv[tid] || (v2 == bv[tid] && i2 < bi[tid])) { bv[tid] = v2; bi[tid] = i2; }
        }
        __syncthreads();
    }
    if (tid == 0) {
        int nidx = bi[0];
        int done = state[1];
        outIdx[0] = (float)(done ? EOSID : nidx);
        if (!done) {
            state[0] = nidx;
            if (nidx == EOSID) state[1] = 1;
        }
    }
}

extern "C" void kernel_launch(void* const* d_in, const int* in_sizes, int n_in,
                              void* d_out, int out_size, void* d_ws, size_t ws_size,
                              hipStream_t stream) {
    (void)in_sizes; (void)n_in; (void)out_size; (void)ws_size;
    const int*   src    = (const int*)d_in[0];
    // d_in[1] = max_len (device scalar, fixed at 64 by setup)
    const float* encWih = (const float*)d_in[2];
    const float* encWhh = (const float*)d_in[3];
    const float* encBih = (const float*)d_in[4];
    const float* encBhh = (const float*)d_in[5];
    const float* decWih = (const float*)d_in[6];
    const float* decWhh = (const float*)d_in[7];
    const float* decBih = (const float*)d_in[8];
    const float* decBhh = (const float*)d_in[9];
    const float* projW  = (const float*)d_in[10];
    const float* projB  = (const float*)d_in[11];
    float* out = (float*)d_out;

    char* ws = (char*)d_ws;
    size_t o = 0;
    auto take = [&](size_t n) -> char* {
        char* p = ws + o;
        o += (n + 255) & ~(size_t)255;
        return p;
    };
    unsigned short* projBf = (unsigned short*)take((size_t)VPAD * HSZ * 2);
    unsigned short* encBf  = (unsigned short*)take((size_t)GSZ * HSZ * 2);
    unsigned short* decBf  = (unsigned short*)take((size_t)GSZ * HSZ * 2);
    float*          xg     = (float*)take((size_t)128 * GSZ * 4);
    float*          cWs    = (float*)take(HSZ * 4);
    unsigned short* hbfWs  = (unsigned short*)take(HSZ * 2);
    int*            state  = (int*)take(2 * 4);
    float*          pVal   = (float*)take(64 * 4);
    int*            pIdx   = (int*)take(64 * 4);

    convert_kernel<<<2048, 256, 0, stream>>>(encWhh, decWhh, projW, encBf, decBf, projBf);
    xg_kernel<<<(128 * GSZ + 255) / 256, 256, 0, stream>>>(encWih, encBih, src, xg);
    enc_kernel<<<1, 256, 0, stream>>>(xg, encBf, encBhh, cWs, hbfWs, state);
    for (int t = 0; t < 64; ++t) {
        dec_cell_kernel<<<1, 256, 0, stream>>>(decBf, decWih, decBih, decBhh, state, cWs, hbfWs);
        proj_kernel<<<64, 256, 0, stream>>>(projBf, hbfWs, projB, state,
                                            out + 64 + (size_t)t * VTOT, pVal, pIdx);
        argmax_kernel<<<1, 64, 0, stream>>>(pVal, pIdx, state, out + t);
    }
}
